// NetConv_36378372997405
// MI455X (gfx1250) — matrix operand order repaired and checked by reading, not compiled, not run on hardware
//
#include <hip/hip_runtime.h>
#include <hip/hip_bf16.h>

#define IN_CH  16
#define OUT_CH 32
#define EDGE_F 16

typedef __attribute__((ext_vector_type(16))) __bf16 v16bf;
typedef __attribute__((ext_vector_type(8)))  float  v8f;

// float -> bf16 bits, round-to-nearest-even
static __device__ __forceinline__ unsigned short f2bf_bits(float f) {
    unsigned u = __builtin_bit_cast(unsigned, f);
    u = u + 0x7FFFu + ((u >> 16) & 1u);
    return (unsigned short)(u >> 16);
}
static __device__ __forceinline__ __bf16 bits2bf(unsigned short b) {
    return __builtin_bit_cast(__bf16, b);
}

__global__ void __launch_bounds__(256) zero_ws_kernel(float* __restrict__ p, int n) {
    int i = blockIdx.x * blockDim.x + threadIdx.x;
    int stride = gridDim.x * blockDim.x;
    for (; i < n; i += stride) p[i] = 0.0f;
}

// One wave (32 lanes) processes 32 edges per iteration (two 16-edge A tiles).
// Edge MLP via v_wmma_f32_16x16x32_bf16 with bias folded in as an extra K row
// (A has 1.0 at K=16, B's K=16 row holds b_fnn). Second contraction stays fp32.
__global__ void __launch_bounds__(256) nnconv_edge_kernel(
    const float*     __restrict__ x,      // [N, 16]
    const long long* __restrict__ eidx,   // [2, E] int64
    const float*     __restrict__ eattr,  // [E, 16]
    const float*     __restrict__ Wf,     // [16, 512]
    const float*     __restrict__ bfnn,   // [512]
    float*           __restrict__ sums,   // [N, 32]
    float*           __restrict__ cnt,    // [N]
    int E, int numTiles)
{
    // B operand (bf16) pre-arranged in WMMA B layout, one 16-col tile per t:
    //   lane<16 : elems 0..7 = K0..7, elem 8 = K16 (carries b_fnn[c]) of column c=t*16+lane
    //   lane>=16: elems 0..7 = K8..15, elems 8..15 = K24..31 = 0 of column c=t*16+(lane-16)
    __shared__ __align__(32) unsigned short Bs[32 * 32 * 16];  // 32 KB
    __shared__ __align__(32) float xts[8][16 * 32];            // per-wave x[src], [chan][edge] (16 KB)

    const int tid = threadIdx.x;

    for (int slot = tid; slot < 1024; slot += 256) {
        const int t = slot >> 5;
        const int l = slot & 31;
        const int base = slot * 16;
        {
            const int col = t * 16 + (l & 15);
            const int h = l >> 4;
#pragma unroll
            for (int k = 0; k < 8; ++k)
                Bs[base + k] = f2bf_bits(Wf[(h * 8 + k) * 512 + col]);
            Bs[base + 8] = (h == 0) ? f2bf_bits(bfnn[col]) : (unsigned short)0;  // K=16 row = MLP bias
#pragma unroll
            for (int k = 9; k < 16; ++k)
                Bs[base + k] = 0;
        }
    }
    __syncthreads();

    const int lane = tid & 31;
    const int wv   = tid >> 5;
    const int hi   = lane >> 4;   // lane half
    const int m    = lane & 15;   // output column within a 16-col tile
    const int wavesTotal = gridDim.x * 8;
    const int waveId     = blockIdx.x * 8 + wv;

    const __bf16 BF_ONE  = bits2bf(0x3F80);
    const __bf16 BF_ZERO = bits2bf(0);

    for (int tile = waveId; tile < numTiles; tile += wavesTotal) {
        const int base = tile * 32;

        // --- per-lane edge (32 edges per wave iteration) ---
        int eL = base + lane; if (eL >= E) eL = E - 1;
        const int s = (int)eidx[eL];
        const int d = (int)eidx[(size_t)E + eL];

        // --- gather x[src]: each lane copies its edge's row, transposed into LDS ---
        {
            const float4* gx = (const float4*)(x + (size_t)s * IN_CH);
            const float4 r0 = gx[0], r1 = gx[1], r2 = gx[2], r3 = gx[3];
            float* xw = &xts[wv][0];
            xw[ 0 * 32 + lane] = r0.x;  xw[ 1 * 32 + lane] = r0.y;
            xw[ 2 * 32 + lane] = r0.z;  xw[ 3 * 32 + lane] = r0.w;
            xw[ 4 * 32 + lane] = r1.x;  xw[ 5 * 32 + lane] = r1.y;
            xw[ 6 * 32 + lane] = r1.z;  xw[ 7 * 32 + lane] = r1.w;
            xw[ 8 * 32 + lane] = r2.x;  xw[ 9 * 32 + lane] = r2.y;
            xw[10 * 32 + lane] = r2.z;  xw[11 * 32 + lane] = r2.w;
            xw[12 * 32 + lane] = r3.x;  xw[13 * 32 + lane] = r3.y;
            xw[14 * 32 + lane] = r3.z;  xw[15 * 32 + lane] = r3.w;
        }

        // --- two A operands (edges 0..15 and 16..31), WMMA 16-bit A layout ---
        // lanes 0-15: M=lane, elems 0..7 = K0..7, elem 8 = K16 = 1.0 (bias row)
        // lanes16-31: M=lane-16, elems 0..7 = K8..15, elems 8..15 = K24..31 = 0
        v16bf av0, av1;
        {
            int eA = base + m;      if (eA >= E) eA = E - 1;
            int eB = base + 16 + m; if (eB >= E) eB = E - 1;
            const float4* ap = (const float4*)(eattr + (size_t)eA * EDGE_F + hi * 8);
            const float4* bp = (const float4*)(eattr + (size_t)eB * EDGE_F + hi * 8);
            const float4 a0 = ap[0], a1 = ap[1];
            const float4 b0 = bp[0], b1 = bp[1];
            av0[0] = bits2bf(f2bf_bits(a0.x)); av0[1] = bits2bf(f2bf_bits(a0.y));
            av0[2] = bits2bf(f2bf_bits(a0.z)); av0[3] = bits2bf(f2bf_bits(a0.w));
            av0[4] = bits2bf(f2bf_bits(a1.x)); av0[5] = bits2bf(f2bf_bits(a1.y));
            av0[6] = bits2bf(f2bf_bits(a1.z)); av0[7] = bits2bf(f2bf_bits(a1.w));
            av1[0] = bits2bf(f2bf_bits(b0.x)); av1[1] = bits2bf(f2bf_bits(b0.y));
            av1[2] = bits2bf(f2bf_bits(b0.z)); av1[3] = bits2bf(f2bf_bits(b0.w));
            av1[4] = bits2bf(f2bf_bits(b1.x)); av1[5] = bits2bf(f2bf_bits(b1.y));
            av1[6] = bits2bf(f2bf_bits(b1.z)); av1[7] = bits2bf(f2bf_bits(b1.w));
            const __bf16 k16 = (hi == 0) ? BF_ONE : BF_ZERO;
            av0[8] = k16; av1[8] = k16;
#pragma unroll
            for (int e2 = 9; e2 < 16; ++e2) { av0[e2] = BF_ZERO; av1[e2] = BF_ZERO; }
        }

        float accA0[8] = {0,0,0,0,0,0,0,0};  // edges 0..15, cols 0..15
        float accA1[8] = {0,0,0,0,0,0,0,0};  // edges 0..15, cols 16..31
        float accB0[8] = {0,0,0,0,0,0,0,0};  // edges 16..31, cols 0..15
        float accB1[8] = {0,0,0,0,0,0,0,0};  // edges 16..31, cols 16..31

        // 16 input channels; per channel two 16-col tiles (t=2i, 2i+1), each B
        // tile reused by both A tiles -> 4 WMMAs per i, issued back-to-back so
        // the ReLU/FMA blocks cover the WMMA->VALU hazard windows (no v_nops).
#pragma unroll 2
        for (int i = 0; i < 16; ++i) {
            const v16bf bv0 = *(const v16bf*)&Bs[((2 * i)     * 32 + lane) * 16];
            const v16bf bv1 = *(const v16bf*)&Bs[((2 * i + 1) * 32 + lane) * 16];

            v8f c1 = {}, c2 = {}, c3 = {}, c4 = {};
            c1 = __builtin_amdgcn_wmma_f32_16x16x32_bf16(false, av0, false, bv0, (short)0, c1, false, false);
            c2 = __builtin_amdgcn_wmma_f32_16x16x32_bf16(false, av1, false, bv0, (short)0, c2, false, false);
            c3 = __builtin_amdgcn_wmma_f32_16x16x32_bf16(false, av0, false, bv1, (short)0, c3, false, false);
            c4 = __builtin_amdgcn_wmma_f32_16x16x32_bf16(false, av1, false, bv1, (short)0, c4, false, false);

            const float4* xpA = (const float4*)&xts[wv][i * 32 + hi * 8];
            const float4* xpB = (const float4*)&xts[wv][i * 32 + 16 + hi * 8];
            const float4 xa0 = xpA[0], xa1 = xpA[1];
            const float4 xb0 = xpB[0], xb1 = xpB[1];
            const float xvA[8] = {xa0.x, xa0.y, xa0.z, xa0.w, xa1.x, xa1.y, xa1.z, xa1.w};
            const float xvB[8] = {xb0.x, xb0.y, xb0.z, xb0.w, xb1.x, xb1.y, xb1.z, xb1.w};

#pragma unroll
            for (int r = 0; r < 8; ++r) accA0[r] += xvA[r] * fmaxf(c1[r], 0.0f);
#pragma unroll
            for (int r = 0; r < 8; ++r) accB0[r] += xvB[r] * fmaxf(c2[r], 0.0f);
#pragma unroll
            for (int r = 0; r < 8; ++r) accA1[r] += xvA[r] * fmaxf(c3[r], 0.0f);
#pragma unroll
            for (int r = 0; r < 8; ++r) accB1[r] += xvB[r] * fmaxf(c4[r], 0.0f);
        }

        // --- scatter-add into target nodes (hardware f32 atomics, L2-resident) ---
#pragma unroll
        for (int r = 0; r < 8; ++r) {
            const int eoA = r + hi * 8;        // edge-in-tile 0..15
            const int eoB = 16 + r + hi * 8;   // edge-in-tile 16..31
            const int dA = __shfl(d, eoA, 32);
            const int dB = __shfl(d, eoB, 32);
            if (base + eoA < E) {
                unsafeAtomicAdd(&sums[(size_t)dA * OUT_CH + m],      accA0[r]);
                unsafeAtomicAdd(&sums[(size_t)dA * OUT_CH + 16 + m], accA1[r]);
                if (m == 0) unsafeAtomicAdd(&cnt[dA], 1.0f);
            }
            if (base + eoB < E) {
                unsafeAtomicAdd(&sums[(size_t)dB * OUT_CH + m],      accB0[r]);
                unsafeAtomicAdd(&sums[(size_t)dB * OUT_CH + 16 + m], accB1[r]);
                if (m == 0) unsafeAtomicAdd(&cnt[dB], 1.0f);
            }
        }
    }
}

// out[n,o] = sums[n,o]/max(cnt,1) + x[n,:]@root[:,o] + bias[o]
__global__ void __launch_bounds__(256) finalize_kernel(
    const float* __restrict__ x, const float* __restrict__ root,
    const float* __restrict__ bias, const float* __restrict__ sums,
    const float* __restrict__ cnt, float* __restrict__ out, int nNodes)
{
    const int t = blockIdx.x * blockDim.x + threadIdx.x;
    if (t >= nNodes * OUT_CH) return;
    const int node = t >> 5;
    const int o    = t & 31;
    float r = sums[t] / fmaxf(cnt[node], 1.0f) + bias[o];
    const float* xr = x + (size_t)node * IN_CH;
#pragma unroll
    for (int i = 0; i < IN_CH; ++i)
        r += xr[i] * root[i * OUT_CH + o];
    out[t] = r;
}

extern "C" void kernel_launch(void* const* d_in, const int* in_sizes, int n_in,
                              void* d_out, int out_size, void* d_ws, size_t ws_size,
                              hipStream_t stream) {
    const float*     x     = (const float*)d_in[0];
    const long long* eidx  = (const long long*)d_in[1];
    const float*     eattr = (const float*)d_in[2];
    const float*     Wf    = (const float*)d_in[3];
    const float*     bfnn  = (const float*)d_in[4];
    const float*     root  = (const float*)d_in[5];
    const float*     bias  = (const float*)d_in[6];
    float* out = (float*)d_out;

    const int nNodes   = in_sizes[0] / IN_CH;
    const int E        = in_sizes[1] / 2;
    const int numTiles = (E + 31) / 32;

    float* sums = (float*)d_ws;                       // [nNodes, 32]
    float* cnt  = sums + (size_t)nNodes * OUT_CH;     // [nNodes]

    const int zn = nNodes * (OUT_CH + 1);
    zero_ws_kernel<<<2048, 256, 0, stream>>>(sums, zn);

    nnconv_edge_kernel<<<640, 256, 0, stream>>>(
        x, eidx, eattr, Wf, bfnn, sums, cnt, E, numTiles);

    const int ft = nNodes * OUT_CH;
    finalize_kernel<<<(ft + 255) / 256, 256, 0, stream>>>(
        x, root, bias, sums, cnt, out, nNodes);
}
